// Network_1400159338851
// MI455X (gfx1250) — compile-verified, matrix operand order checked by reading, not run on hardware
//
#include <hip/hip_runtime.h>

// ---------------------------------------------------------------------------
// CDNA5 (gfx1250) spiking CNN. Time-major activations [t][neuron]:
//   - LIF scan loads/stores fully coalesced (dominant HBM traffic)
//   - FC A-tiles contiguous in K -> vectorized 16B loads
// Weights staged in LDS via CDNA5 async global->LDS (ASYNCcnt) when available.
// ---------------------------------------------------------------------------

typedef __attribute__((ext_vector_type(16))) __bf16       v16bf;
typedef __attribute__((ext_vector_type(8)))  float        v8f;
typedef __attribute__((ext_vector_type(4)))  unsigned int v4u;
typedef __attribute__((ext_vector_type(4)))  int          v4i;

#define T_STEPS 128
#define BATCH   8
#define THETA_F   5120.0f   // 80 * 64
#define DECAY_U_F 0.75f     // 1 - 1024/4096
#define DECAY_V_F 0.96875f  // 1 - 128/4096
#define POOL_W_F  88.0f     // 1.1 * 80

#if __has_builtin(__builtin_amdgcn_global_load_async_to_lds_b128) && \
    __has_builtin(__builtin_amdgcn_s_wait_asynccnt)
#define USE_ASYNC_LDS 1
#else
#define USE_ASYNC_LDS 0
#endif

typedef __attribute__((address_space(1))) v4i* gv4i_p;   // global
typedef __attribute__((address_space(3))) v4i* lv4i_p;   // LDS

// Stage one 16B chunk global -> LDS (async path bypasses VGPRs; tracked by
// ASYNCcnt instead of LOADcnt/DScnt).
__device__ __forceinline__ void stage16(const __bf16* g, __bf16* l) {
#if USE_ASYNC_LDS
  __builtin_amdgcn_global_load_async_to_lds_b128((gv4i_p)g, (lv4i_p)l, 0, 0);
#else
  *(v4u*)l = *(const v4u*)g;
#endif
}
__device__ __forceinline__ void stage_fence() {
#if USE_ASYNC_LDS
  __builtin_amdgcn_s_wait_asynccnt(0);
#endif
  __syncthreads();
}

// ---- WMMA lane/K mapping for 16-bit A/B 16x32 tiles (ISA 7.12.2) ----------
__device__ __forceinline__ constexpr int kmap(int e, int hi) {
  return (e < 8) ? (hi * 8 + e) : (16 + hi * 8 + (e - 8));
}

__device__ __forceinline__ v8f zero_v8f() {
  v8f z;
#pragma unroll
  for (int i = 0; i < 8; ++i) z[i] = 0.0f;
  return z;
}

// Row-major [rows][ld] bf16 tile fetch: two contiguous 16B runs per lane.
template <typename P>
__device__ __forceinline__ v16bf load_tile_rm(P base, int row, int k0, int ld, int lane) {
  int hi = lane >> 4;
  P p = base + (size_t)row * ld + k0 + hi * 8;
  union { v16bf v; v4u q[2]; } u;
  u.q[0] = *(const v4u*)(p);
  u.q[1] = *(const v4u*)(p + 16);
  return u.v;
}

// ---------------------------------------------------------------------------
// Weight packing: f32 [Nreal, Kreal] -> bf16 [Npad, Kpad] zero-padded
// ---------------------------------------------------------------------------
__global__ void pack_weights(const float* __restrict__ src, __bf16* __restrict__ dst,
                             int Nreal, int Kreal, int Npad, int Kpad) {
  int i = blockIdx.x * blockDim.x + threadIdx.x;
  if (i >= Npad * Kpad) return;
  int n = i / Kpad, k = i % Kpad;
  float v = (n < Nreal && k < Kreal) ? src[n * Kreal + k] : 0.0f;
  dst[i] = (__bf16)v;
}

// ---------------------------------------------------------------------------
// 4x4 sum-pool on raw input x [B,2,128,128,T] f32 (time-last, harness layout)
// -> drive time-major [t][(b*2+c)*1024 + h*32 + w] f32
// ---------------------------------------------------------------------------
__global__ void pool4_kernel(const float* __restrict__ x, float* __restrict__ drive) {
  constexpr int NOUT = 16 * 32 * 32;                 // 16384 neurons
  int i = blockIdx.x * blockDim.x + threadIdx.x;     // i = t*NOUT + n
  int n = i & (NOUT - 1);
  int t = i >> 14;
  int w = n & 31;
  int h = (n >> 5) & 31;
  int bc = n >> 10;                                  // b*2 + c
  const float* px = x + (((size_t)bc * 128 + h * 4) * 128 + w * 4) * T_STEPS + t;
  float s = 0.0f;
#pragma unroll
  for (int dh = 0; dh < 4; ++dh)
#pragma unroll
    for (int dw = 0; dw < 4; ++dw)
      s += px[((size_t)dh * 128 + dw) * T_STEPS];
  drive[i] = s * POOL_W_F;
}

// ---------------------------------------------------------------------------
// 2x2 sum-pool on time-major bf16 spikes [t][bc*HIN*HIN + h*HIN + w]
// -> drive [t][bc*HO*HO + ho*HO + wo] f32
// ---------------------------------------------------------------------------
template <int CHANS, int HIN>
__global__ void pool2_kernel(const __bf16* __restrict__ s_in, float* __restrict__ drive) {
  constexpr int HO = HIN / 2;
  constexpr int NOUT = CHANS * HO * HO;
  int i = blockIdx.x * blockDim.x + threadIdx.x;     // i = t*NOUT + n
  int n = i % NOUT;
  int t = i / NOUT;
  int wo = n % HO;
  int ho = (n / HO) % HO;
  int bc = n / (HO * HO);
  const __bf16* p = s_in + (size_t)t * (CHANS * HIN * HIN)
                  + (size_t)bc * (HIN * HIN) + (ho * 2) * HIN + wo * 2;
  float s = (float)p[0] + (float)p[1] + (float)p[HIN] + (float)p[HIN + 1];
  drive[i] = s * POOL_W_F;
}

// ---------------------------------------------------------------------------
// Loihi CUBA LIF scan; time-major drive/spikes, delay-shift fused:
// spikes[0][n] = 0 ; spikes[t+1][n] = spike at step t. Fully coalesced.
// ---------------------------------------------------------------------------
__global__ void lif_kernel(const float* __restrict__ drive, __bf16* __restrict__ spikes,
                           int N) {
  int n = blockIdx.x * blockDim.x + threadIdx.x;
  if (n >= N) return;
  float u = 0.0f, v = 0.0f, ref = 0.0f;
  spikes[n] = (__bf16)0.0f;                          // t = 0 after shift
  for (int t = 0; t < T_STEPS; ++t) {
    u = u * DECAY_U_F + drive[(size_t)t * N + n] * 64.0f;
    float active = (ref <= 0.0f) ? 1.0f : 0.0f;
    v = (v * DECAY_V_F + u) * active;
    float sp = (v >= THETA_F) ? 1.0f : 0.0f;
    ref = (sp > 0.0f) ? 1.0f : fmaxf(ref - 1.0f, 0.0f);
    v = v * (1.0f - sp);
    if (t + 1 < T_STEPS) spikes[(size_t)(t + 1) * N + n] = (__bf16)sp;
  }
}

// ---------------------------------------------------------------------------
// Implicit-GEMM conv ('SAME') via bf16 WMMA, weights cached in LDS.
//  rows m = t*(B*H*W) + ((b*H+h)*W+w) ; K = (c*KH+kh)*KW+kw  (zero padded)
//  NT = N-tiles per wave (reuses the im2col A-gather). Fully unrolled K.
// ---------------------------------------------------------------------------
template <int CIN, int H, int W, int O, int KH, int KW, int PAD,
          int KREAL, int KPAD, int NT>
__global__ void conv_wmma_kernel(const __bf16* __restrict__ S,
                                 const __bf16* __restrict__ Wp,
                                 float* __restrict__ Dout) {
  constexpr int PIXT = BATCH * H * W;     // pixels per timestep (power of 2)
  constexpr int NIN  = BATCH * CIN * H * W;
  constexpr int NOUTT = BATCH * O * H * W;

  __shared__ __bf16 wsh[O * KPAD];
  {
    constexpr int CHUNKS = (O * KPAD) / 8;          // 16B chunks
    for (int c = threadIdx.x; c < CHUNKS; c += blockDim.x)
      stage16(Wp + c * 8, &wsh[c * 8]);
  }
  stage_fence();

  int wave = (blockIdx.x * blockDim.x + threadIdx.x) >> 5;  // == tile-M index
  int lane = threadIdx.x & 31;
  int hi = lane >> 4;
  int m0 = wave * 16;

  // decode A row for this lane (all constexpr strides -> shifts/masks)
  int mr = m0 + (lane & 15);
  int t = mr / PIXT;
  int pix = mr % PIXT;
  int wq = pix % W;
  int hq = (pix / W) % H;
  int b = pix / (W * H);

  v8f acc[NT];
#pragma unroll
  for (int nt = 0; nt < NT; ++nt) acc[nt] = zero_v8f();

#pragma unroll
  for (int k0 = 0; k0 < KPAD; k0 += 32) {
    v16bf a;
#pragma unroll
    for (int e = 0; e < 16; ++e) {
      int k = k0 + kmap(e, hi);
      __bf16 val = (__bf16)0.0f;
      if (k < KREAL) {
        int kw = k % KW;
        int kh = (k / KW) % KH;
        int ci = k / (KW * KH);
        int ih = hq + kh - PAD, iw = wq + kw - PAD;
        if ((unsigned)ih < (unsigned)H && (unsigned)iw < (unsigned)W)
          val = S[(size_t)t * NIN + ((b * CIN + ci) * H + ih) * W + iw];
      }
      a[e] = val;
    }
#pragma unroll
    for (int nt = 0; nt < NT; ++nt) {
      v16bf bt = load_tile_rm(&wsh[0], nt * 16 + (lane & 15), k0, KPAD, lane);
      acc[nt] = __builtin_amdgcn_wmma_f32_16x16x32_bf16(false, a, false, bt,
                                                        (short)0, acc[nt], false, false);
    }
  }

#pragma unroll
  for (int nt = 0; nt < NT; ++nt) {
    int n = nt * 16 + (lane & 15);
#pragma unroll
    for (int v = 0; v < 8; ++v) {
      int m = m0 + hi * 8 + v;
      int tt = m / PIXT;
      int pp = m % PIXT;
      int ww = pp % W;
      int hh = (pp / W) % H;
      int bb = pp / (W * H);
      Dout[(size_t)tt * NOUTT + ((bb * O + n) * H + hh) * W + ww] = acc[nt][v];
    }
  }
}

// ---------------------------------------------------------------------------
// FC layer via bf16 WMMA. A = spikes [t][b][f] == row-major [M=T*B][F].
// B-panel (16 rows x F) staged in dynamic LDS (async), shared by 8 waves;
// each wave computes a pair of M-tiles reusing the LDS B-tile.
// grid = (blocksPerN = M/(32*8), Ntiles) ; block = 256.
// ---------------------------------------------------------------------------
template <int F, int OREAL>
__global__ void fc_wmma_kernel(const __bf16* __restrict__ S,
                               const __bf16* __restrict__ Wp,
                               float* __restrict__ Dout) {
  extern __shared__ __bf16 bsh[];                    // [16][F]
  int n0g = blockIdx.y * 16;
  {
    constexpr int CHUNKS = (16 * F) / 8;
    const __bf16* wsrc = Wp + (size_t)n0g * F;
    for (int c = threadIdx.x; c < CHUNKS; c += blockDim.x)
      stage16(wsrc + c * 8, &bsh[c * 8]);
  }
  stage_fence();

  int lane = threadIdx.x & 31;
  int pair = blockIdx.x * 8 + (threadIdx.x >> 5);
  int m0 = pair * 32;                                // two tiles: m0, m0+16

  v8f acc0 = zero_v8f(), acc1 = zero_v8f();
#pragma unroll 4
  for (int k0 = 0; k0 < F; k0 += 32) {
    if (k0 + 128 < F)                                // stream A ahead
      __builtin_prefetch((const void*)(S + (size_t)(m0 + (lane & 15)) * F + k0 + 128), 0, 0);
    v16bf bt = load_tile_rm(&bsh[0], lane & 15, k0, F, lane);
    v16bf a0 = load_tile_rm(S, m0 + (lane & 15), k0, F, lane);
    v16bf a1 = load_tile_rm(S, m0 + 16 + (lane & 15), k0, F, lane);
    acc0 = __builtin_amdgcn_wmma_f32_16x16x32_bf16(false, a0, false, bt,
                                                   (short)0, acc0, false, false);
    acc1 = __builtin_amdgcn_wmma_f32_16x16x32_bf16(false, a1, false, bt,
                                                   (short)0, acc1, false, false);
  }

  int n = n0g + (lane & 15);
  if (n < OREAL) {
    int hi = lane >> 4;
#pragma unroll
    for (int v = 0; v < 8; ++v) {
      Dout[(size_t)(m0 + hi * 8 + v) * OREAL + n] = acc0[v];
      Dout[(size_t)(m0 + 16 + hi * 8 + v) * OREAL + n] = acc1[v];
    }
  }
}

// final spikes s6 time-major [t][b*11+o] -> out [b][o][t] f32
__global__ void cast_out_kernel(const __bf16* __restrict__ s, float* __restrict__ out,
                                int total) {
  int i = blockIdx.x * blockDim.x + threadIdx.x;     // i = (b*11+o)*T + t
  if (i >= total) return;
  int t = i & (T_STEPS - 1);
  int r = i >> 7;
  out[i] = (float)s[(size_t)t * 88 + r];
}

// ---------------------------------------------------------------------------
extern "C" void kernel_launch(void* const* d_in, const int* in_sizes, int n_in,
                              void* d_out, int out_size, void* d_ws, size_t ws_size,
                              hipStream_t stream) {
  const float* x   = (const float*)d_in[0];  // [8,2,128,128,128]
  const float* w1  = (const float*)d_in[1];  // [16,2,5,5]
  const float* w2  = (const float*)d_in[2];  // [32,16,3,3]
  const float* wf1 = (const float*)d_in[3];  // [512,2048]
  const float* wf2 = (const float*)d_in[4];  // [11,512]
  float* out = (float*)d_out;                // [8,11,1,1,128]

  char* ws = (char*)d_ws;
  size_t off = 0;
  auto carve = [&](size_t bytes) -> void* {
    void* p = (void*)(ws + off);
    off = (off + bytes + 255) & ~(size_t)255;
    return p;
  };
  float*  dbuf = (float*)carve((size_t)8 * 16 * 32 * 32 * T_STEPS * 4);  // 67 MB max drive
  __bf16* sA   = (__bf16*)carve((size_t)8 * 16 * 32 * 32 * T_STEPS * 2); // 33.5 MB
  __bf16* sB   = (__bf16*)carve((size_t)8 * 16 * 32 * 32 * T_STEPS * 2); // 33.5 MB
  __bf16* W1p  = (__bf16*)carve((size_t)16 * 64 * 2);
  __bf16* W2p  = (__bf16*)carve((size_t)32 * 160 * 2);
  __bf16* Wf1p = (__bf16*)carve((size_t)512 * 2048 * 2);
  __bf16* Wf2p = (__bf16*)carve((size_t)16 * 512 * 2);

  pack_weights<<<(16 * 64 + 255) / 256, 256, 0, stream>>>(w1, W1p, 16, 50, 16, 64);
  pack_weights<<<(32 * 160 + 255) / 256, 256, 0, stream>>>(w2, W2p, 32, 144, 32, 160);
  pack_weights<<<(512 * 2048 + 255) / 256, 256, 0, stream>>>(wf1, Wf1p, 512, 2048, 512, 2048);
  pack_weights<<<(16 * 512 + 255) / 256, 256, 0, stream>>>(wf2, Wf2p, 11, 512, 16, 512);

  // L0: pool4 + LIF -> s0 [t][16384]
  pool4_kernel<<<(16384 * T_STEPS) / 256, 256, 0, stream>>>(x, dbuf);
  lif_kernel<<<16384 / 256, 256, 0, stream>>>(dbuf, sA, 16384);

  // L1: conv1 (M=1,048,576 N=16 K=64, NT=1) + LIF -> s1 [t][131072]
  conv_wmma_kernel<2, 32, 32, 16, 5, 5, 2, 50, 64, 1>
      <<<(65536 * 32) / 256, 256, 0, stream>>>(sA, W1p, dbuf);
  lif_kernel<<<131072 / 256, 256, 0, stream>>>(dbuf, sB, 131072);

  // L2: pool2 + LIF -> s2 [t][32768]
  pool2_kernel<8 * 16, 32><<<(32768 * T_STEPS) / 256, 256, 0, stream>>>(sB, dbuf);
  lif_kernel<<<32768 / 256, 256, 0, stream>>>(dbuf, sA, 32768);

  // L3: conv2 (M=262,144 N=32 K=160, NT=2) + LIF -> s3 [t][65536]
  conv_wmma_kernel<16, 16, 16, 32, 3, 3, 1, 144, 160, 2>
      <<<(16384 * 32) / 256, 256, 0, stream>>>(sA, W2p, dbuf);
  lif_kernel<<<65536 / 256, 256, 0, stream>>>(dbuf, sB, 65536);

  // L4: pool2 + LIF -> s4 [t][b][2048]
  pool2_kernel<8 * 32, 16><<<(16384 * T_STEPS) / 256, 256, 0, stream>>>(sB, dbuf);
  lif_kernel<<<16384 / 256, 256, 0, stream>>>(dbuf, sA, 16384);

  // L5: fc1 (M=1024 N=512 K=2048) + LIF -> s5 [t][b][512]
  {
    dim3 grid(1024 / (32 * 8), 512 / 16);            // (4, 32)
    fc_wmma_kernel<2048, 512><<<grid, 256, 16 * 2048 * 2, stream>>>(sA, Wf1p, dbuf);
    lif_kernel<<<4096 / 256, 256, 0, stream>>>(dbuf, sB, 4096);
  }
  // L6: fc2 (M=1024 N=16(11) K=512) + LIF -> s6 [t][88] (already delayed)
  {
    dim3 grid(1024 / (32 * 8), 1);                   // (4, 1)
    fc_wmma_kernel<512, 11><<<grid, 256, 16 * 512 * 2, stream>>>(sB, Wf2p, dbuf);
    lif_kernel<<<1, 256, 0, stream>>>(dbuf, sA, 88);
    cast_out_kernel<<<(8 * 11 * T_STEPS + 255) / 256, 256, 0, stream>>>(sA, out, 8 * 11 * T_STEPS);
  }
  (void)in_sizes; (void)n_in; (void)out_size; (void)ws_size;
}
